// GraphAutoencoderNeighborhood_43001212567666
// MI455X (gfx1250) — matrix operand order repaired and checked by reading, not compiled, not run on hardware
//
#include <hip/hip_runtime.h>
#include <cstdint>

// ---------------------------------------------------------------------------
// Problem constants (from reference)
// ---------------------------------------------------------------------------
#define NNODES 100000
#define NEDGES 1600000
#define EPRIME (NEDGES + NNODES)   // with self loops
#define EPS_BN 1e-5f
#define K0PAD 160                  // enc0 input dim 136 padded to multiple of 32

typedef __attribute__((ext_vector_type(16))) __bf16 v16bf;
typedef __attribute__((ext_vector_type(8)))  float  v8f;

// ---------------------------------------------------------------------------
// WMMA GEMM: Y[N,M] = act( X[N,K] @ W[K,M] + bias )
// Preconditions: N % 16 == 0, K % 32 == 0, M % (16*WAVES) == 0 per launch.
// WAVES waves per block; block tile = 16 rows x 16*WAVES cols (full tiles, no
// guards). A tile staged by async LDS-DMA (global_load_async_to_lds_b128,
// ASYNCcnt), double buffered; B tile register-prefetched one k-step ahead and
// transposed into LDS. bf16 inputs, f32 accumulate: v_wmma_f32_16x16x32_bf16.
// ---------------------------------------------------------------------------
template <int WAVES>
__global__ __launch_bounds__(32 * WAVES)
void gemm_wmma_kernel(const float* __restrict__ X, const float* __restrict__ W,
                      const float* __restrict__ bias, float* __restrict__ Y,
                      int N, int K, int M, int act /*0:none 1:relu*/)
{
    __shared__ alignas(16) float As[2][16][32];            // X tile (dbl buf)
    __shared__ alignas(16) float Bs[2][16 * WAVES][36];    // W^T tile (dbl buf)

    const int t    = threadIdx.x;
    const int wave = t >> 5;
    const int lane = t & 31;
    const int half = lane >> 4;     // 0: lanes 0-15, 1: lanes 16-31
    const int l16  = lane & 15;

    const int row0  = blockIdx.y * 16;
    const int col0b = blockIdx.x * (16 * WAVES);
    const int mloc  = wave * 16 + l16;
    const int col   = col0b + mloc;

    constexpr int NT  = 32 * WAVES;  // threads per block
    constexpr int CH  = 4 / WAVES;   // A 16B-chunks per thread (W4:1, W2:2)
    constexpr int GPR = 4 * WAVES;   // B float4 groups per k-row

    // issue async DMA of A tile (16x32 floats) for k-offset kt into buffer p
    auto issueA = [&](int kt, int p) {
#pragma unroll
        for (int j = 0; j < CH; ++j) {
            int ci = t + j * NT;           // chunk id 0..127
            int r  = ci >> 3;              // row 0..15
            int kk = (ci & 7) * 4;         // col 0..28
            unsigned lo = (unsigned)(uintptr_t)&As[p][r][kk];  // LDS byte off
            const float* gp = &X[(size_t)(row0 + r) * K + kt + kk];
            asm volatile("global_load_async_to_lds_b128 %0, %1, off"
                         :: "v"(lo), "v"(gp) : "memory");
        }
    };

    float4 Breg[4];
    auto loadB = [&](int kt) {
#pragma unroll
        for (int j = 0; j < 4; ++j) {
            int f4 = t + j * NT;           // float4 id within 32 x (16*WAVES)
            int k  = f4 / GPR;
            int m4 = (f4 % GPR) * 4;
            Breg[j] = *(const float4*)&W[(size_t)(kt + k) * M + col0b + m4];
        }
    };

    issueA(0, 0);
    loadB(0);

    v8f acc = {};
    int p = 0;
    for (int k0 = 0; k0 < K; k0 += 32, p ^= 1) {
        const bool has_next = (k0 + 32) < K;

        // (a) my DMA chunks for As[p] are complete
        asm volatile("s_wait_asynccnt 0x0" ::: "memory");

        // (b) commit prefetched B regs, transposed, into Bs[p]
#pragma unroll
        for (int j = 0; j < 4; ++j) {
            int f4 = t + j * NT;
            int k  = f4 / GPR;
            int m4 = (f4 % GPR) * 4;
            Bs[p][m4 + 0][k] = Breg[j].x;
            Bs[p][m4 + 1][k] = Breg[j].y;
            Bs[p][m4 + 2][k] = Breg[j].z;
            Bs[p][m4 + 3][k] = Breg[j].w;
        }
        __syncthreads();                                   // (c) publish tile p

        // (d) prefetch next tile into buffer p^1 / registers (overlaps compute)
        if (has_next) {
            issueA(k0 + 32, p ^ 1);
            loadB(k0 + 32);
        }

        // (e) fragments from LDS
        // A (ISA 16-bit A 16x32): lane -> row l16; e=0..7 -> K=half*8+e,
        // e=8..15 -> K=16+half*8+(e-8): two contiguous 8-float runs.
        float af[16], bfv[16];
        *(float4*)&af[0]  = *(const float4*)&As[p][l16][half * 8];
        *(float4*)&af[4]  = *(const float4*)&As[p][l16][half * 8 + 4];
        *(float4*)&af[8]  = *(const float4*)&As[p][l16][16 + half * 8];
        *(float4*)&af[12] = *(const float4*)&As[p][l16][16 + half * 8 + 4];
        // B: lane -> column mloc; same K order as A (two contiguous 8-float runs).
        *(float4*)&bfv[0]  = *(const float4*)&Bs[p][mloc][half * 8];
        *(float4*)&bfv[4]  = *(const float4*)&Bs[p][mloc][half * 8 + 4];
        *(float4*)&bfv[8]  = *(const float4*)&Bs[p][mloc][16 + half * 8];
        *(float4*)&bfv[12] = *(const float4*)&Bs[p][mloc][16 + half * 8 + 4];

        v16bf a, b;
#pragma unroll
        for (int e = 0; e < 16; ++e) {
            a[e] = (__bf16)af[e];
            b[e] = (__bf16)bfv[e];
        }

        acc = __builtin_amdgcn_wmma_f32_16x16x32_bf16(
            false, a, false, b, (short)0, acc, false, false);

        __syncthreads();   // (f) reads of tile p done before it is re-DMA'd
    }

    // C/D layout: lane -> column l16; VGPR r -> row r + half*8
    float bv = bias ? bias[col] : 0.0f;
#pragma unroll
    for (int r = 0; r < 8; ++r) {
        int row = row0 + r + half * 8;
        float v = acc[r] + bv;
        if (act == 1) v = fmaxf(v, 0.0f);
        Y[(size_t)row * M + col] = v;
    }
}

// ---------------------------------------------------------------------------
// Embedding lookup + concat into padded buffer:
// h0[n] = [ emb[id(n)] (8) | x[n,1:129] (128) | zeros (24) ]   width K0PAD
// ---------------------------------------------------------------------------
__global__ void embed_concat_kernel(const float* __restrict__ x,
                                    const float* __restrict__ emb,
                                    float* __restrict__ h0, int N)
{
    size_t idx = (size_t)blockIdx.x * blockDim.x + threadIdx.x;
    size_t total = (size_t)N * K0PAD;
    if (idx >= total) return;
    int n = (int)(idx / K0PAD);
    int c = (int)(idx % K0PAD);
    float v;
    if (c < 8) {
        int id = (int)x[(size_t)n * 129];
        v = emb[id * 8 + c];
    } else if (c < 136) {
        v = x[(size_t)n * 129 + 1 + (c - 8)];
    } else {
        v = 0.0f;
    }
    h0[idx] = v;
}

// ---------------------------------------------------------------------------
// Per-node attention logits: ssrc[n,h] = sum_c hw[n,h,c]*as[h,c]; same for ad
// ---------------------------------------------------------------------------
__global__ void gat_logits_kernel(const float* __restrict__ hw,
                                  const float* __restrict__ as_,
                                  const float* __restrict__ ad_,
                                  float* __restrict__ ssrc,
                                  float* __restrict__ sdst,
                                  int N, int H, int C)
{
    int idx = blockIdx.x * blockDim.x + threadIdx.x;
    if (idx >= N * H) return;
    int n = idx / H, h = idx % H;
    const float* hp = hw + (size_t)n * H * C + (size_t)h * C;
    const float* ap = as_ + (size_t)h * C;
    const float* dp = ad_ + (size_t)h * C;
    float s = 0.f, d = 0.f;
    for (int c = 0; c < C; ++c) { float v = hp[c]; s += v * ap[c]; d += v * dp[c]; }
    ssrc[idx] = s;
    sdst[idx] = d;
}

// ordered-uint mapping for float atomic max (init value 0u == -inf)
__device__ __forceinline__ unsigned f2o(float f) {
    unsigned u = __float_as_uint(f);
    return (u >> 31) ? ~u : (u | 0x80000000u);
}
__device__ __forceinline__ float o2f(unsigned u) {
    return (u >> 31) ? __uint_as_float(u & 0x7FFFFFFFu) : __uint_as_float(~u);
}

__global__ void fill_f32_kernel(float* p, float v, size_t n)
{
    size_t i = (size_t)blockIdx.x * blockDim.x + threadIdx.x;
    if (i < n) p[i] = v;
}
__global__ void fill_u32_kernel(unsigned* p, unsigned v, size_t n)
{
    size_t i = (size_t)blockIdx.x * blockDim.x + threadIdx.x;
    if (i < n) p[i] = v;
}

__device__ __forceinline__ void edge_nodes(const int* src, const int* dst,
                                           int E, size_t e, int& s, int& d)
{
    if (e < (size_t)E) { s = src[e]; d = dst[e]; }
    else               { s = (int)(e - E); d = s; }   // self loop
}

// pass 1: segment max of leaky-relu logits over dst
__global__ void edge_max_kernel(const int* __restrict__ src, const int* __restrict__ dst,
                                int E, int N, int H,
                                const float* __restrict__ ssrc,
                                const float* __restrict__ sdst,
                                unsigned* __restrict__ amax)
{
    size_t idx = (size_t)blockIdx.x * blockDim.x + threadIdx.x;
    size_t eh = (size_t)(E + N) * H;
    if (idx >= eh) return;
    size_t e = idx / H; int h = (int)(idx - e * H);
    int s, d; edge_nodes(src, dst, E, e, s, d);
    float a = ssrc[(size_t)s * H + h] + sdst[(size_t)d * H + h];
    a = (a > 0.f) ? a : 0.2f * a;
    atomicMax(&amax[(size_t)d * H + h], f2o(a));
}

// pass 2: ex = exp(a - amax[dst]); den[dst] += ex
__global__ void edge_exp_kernel(const int* __restrict__ src, const int* __restrict__ dst,
                                int E, int N, int H,
                                const float* __restrict__ ssrc,
                                const float* __restrict__ sdst,
                                const unsigned* __restrict__ amax,
                                float* __restrict__ ex, float* __restrict__ den)
{
    size_t idx = (size_t)blockIdx.x * blockDim.x + threadIdx.x;
    size_t eh = (size_t)(E + N) * H;
    if (idx >= eh) return;
    size_t e = idx / H; int h = (int)(idx - e * H);
    int s, d; edge_nodes(src, dst, E, e, s, d);
    float a = ssrc[(size_t)s * H + h] + sdst[(size_t)d * H + h];
    a = (a > 0.f) ? a : 0.2f * a;
    float v = __expf(a - o2f(amax[(size_t)d * H + h]));
    ex[idx] = v;
    atomicAdd(&den[(size_t)d * H + h], v);
}

// pass 3: out[dst, j] += (ex / den[dst]) * hw[src, j]
// One thread per (edge, 4-channel group); C % 4 == 0 so head is uniform.
__global__ void edge_agg_kernel(const int* __restrict__ src, const int* __restrict__ dst,
                                int E, int N, int H, int C,
                                const float* __restrict__ hw,
                                const float* __restrict__ ex,
                                const float* __restrict__ den,
                                float* __restrict__ out)
{
    const int HC = H * C;
    const int G  = HC >> 2;                 // 4-channel groups per edge
    size_t idx = (size_t)blockIdx.x * blockDim.x + threadIdx.x;
    size_t total = (size_t)(E + N) * G;
    if (idx >= total) return;
    size_t e = idx / G;
    int j = (int)(idx - e * G) * 4;
    int h = j / C;
    int s, d; edge_nodes(src, dst, E, e, s, d);
    float alpha = ex[e * H + h] / (den[(size_t)d * H + h] + 1e-16f);
    const float* hs = hw  + (size_t)s * HC + j;
    float*       od = out + (size_t)d * HC + j;
#pragma unroll
    for (int c = 0; c < 4; ++c)
        atomicAdd(&od[c], alpha * hs[c]);
}

// ---------------------------------------------------------------------------
// BatchNorm (training-mode batch stats, biased variance)
// ---------------------------------------------------------------------------
__global__ void colstats_kernel(const float* __restrict__ X, int N, int M,
                                float* __restrict__ mean, float* __restrict__ var)
{
    __shared__ float ss[256], ss2[256];
    int c = blockIdx.x;
    float s = 0.f, s2 = 0.f;
    for (int i = threadIdx.x; i < N; i += blockDim.x) {
        float v = X[(size_t)i * M + c];
        s += v; s2 += v * v;
    }
    ss[threadIdx.x] = s; ss2[threadIdx.x] = s2;
    __syncthreads();
    for (int off = 128; off > 0; off >>= 1) {
        if (threadIdx.x < off) {
            ss[threadIdx.x]  += ss[threadIdx.x + off];
            ss2[threadIdx.x] += ss2[threadIdx.x + off];
        }
        __syncthreads();
    }
    if (threadIdx.x == 0) {
        float m = ss[0] / (float)N;
        mean[c] = m;
        var[c]  = ss2[0] / (float)N - m * m;
    }
}

__global__ void bn_apply_kernel(const float* __restrict__ X,
                                const float* __restrict__ mean,
                                const float* __restrict__ var,
                                const float* __restrict__ g,
                                const float* __restrict__ b,
                                float* __restrict__ Y, int N, int M, int relu)
{
    size_t i = (size_t)blockIdx.x * blockDim.x + threadIdx.x;
    if (i >= (size_t)N * M) return;
    int c = (int)(i % M);
    float v = (X[i] - mean[c]) * rsqrtf(var[c] + EPS_BN) * g[c] + b[c];
    if (relu) v = fmaxf(v, 0.f);
    Y[i] = v;
}

// ---------------------------------------------------------------------------
// Reparameterization + KL partial reduction
// ---------------------------------------------------------------------------
__global__ void z_kl_kernel(const float* __restrict__ mu, const float* __restrict__ lv,
                            const float* __restrict__ eps, float* __restrict__ z,
                            float* __restrict__ klacc, int total)
{
    __shared__ float sh[256];
    int i = blockIdx.x * blockDim.x + threadIdx.x;
    float loc = 0.f;
    if (i < total) {
        float m = mu[i], l = lv[i];
        z[i] = m + eps[i] * __expf(0.5f * l);
        loc = 1.f + l - m * m - __expf(l);
    }
    sh[threadIdx.x] = loc;
    __syncthreads();
    for (int off = 128; off > 0; off >>= 1) {
        if (threadIdx.x < off) sh[threadIdx.x] += sh[threadIdx.x + off];
        __syncthreads();
    }
    if (threadIdx.x == 0) atomicAdd(klacc, sh[0]);
}

__global__ void kl_finalize_kernel(const float* klacc, float* out, int total)
{
    if (blockIdx.x == 0 && threadIdx.x == 0)
        out[0] = -0.5f * (klacc[0] / (float)total);
}

__global__ void bias_sigmoid_kernel(const float* __restrict__ X,
                                    const float* __restrict__ b,
                                    float* __restrict__ Y, int N, int M)
{
    size_t i = (size_t)blockIdx.x * blockDim.x + threadIdx.x;
    if (i >= (size_t)N * M) return;
    int c = (int)(i % M);
    float v = X[i] + b[c];
    Y[i] = 1.f / (1.f + __expf(-v));
}

// ---------------------------------------------------------------------------
// Host orchestration
// ---------------------------------------------------------------------------
static inline unsigned cdiv(size_t a, size_t b) { return (unsigned)((a + b - 1) / b); }

struct GatP { const float* W; const float* as_; const float* ad_; };

static void run_gemm(const float* X, const float* W, const float* bias, float* Y,
                     int K, int M, int act, hipStream_t s)
{
    if (M % 64 == 0) {
        dim3 grid(M / 64, NNODES / 16);
        gemm_wmma_kernel<4><<<grid, 128, 0, s>>>(X, W, bias, Y, NNODES, K, M, act);
    } else {            // M == 32
        dim3 grid(M / 32, NNODES / 16);
        gemm_wmma_kernel<2><<<grid, 64, 0, s>>>(X, W, bias, Y, NNODES, K, M, act);
    }
}

static void run_gat(const float* in, int din, int H, int C, const GatP& p,
                    const int* src, const int* dst,
                    float* hw, float* agg, float* ssrc, float* sdst,
                    unsigned* amax, float* den, float* ex, hipStream_t s)
{
    const int N = NNODES, E = NEDGES;
    const int HC = H * C;
    run_gemm(in, p.W, nullptr, hw, din, HC, 0, s);

    int nh = N * H;
    gat_logits_kernel<<<cdiv(nh, 256), 256, 0, s>>>(hw, p.as_, p.ad_, ssrc, sdst, N, H, C);
    fill_u32_kernel<<<cdiv(nh, 256), 256, 0, s>>>(amax, 0u, (size_t)nh);
    fill_f32_kernel<<<cdiv(nh, 256), 256, 0, s>>>(den, 0.f, (size_t)nh);

    size_t eh = (size_t)EPRIME * H;
    edge_max_kernel<<<cdiv(eh, 256), 256, 0, s>>>(src, dst, E, N, H, ssrc, sdst, amax);
    edge_exp_kernel<<<cdiv(eh, 256), 256, 0, s>>>(src, dst, E, N, H, ssrc, sdst, amax, ex, den);

    size_t nm = (size_t)N * HC;
    fill_f32_kernel<<<cdiv(nm, 256), 256, 0, s>>>(agg, 0.f, nm);
    size_t tot = (size_t)EPRIME * (HC >> 2);
    edge_agg_kernel<<<cdiv(tot, 256), 256, 0, s>>>(src, dst, E, N, H, C, hw, ex, den, agg);
}

static void run_bn(const float* X, const float* g, const float* b, float* Y,
                   int M, int relu, float* mean, float* var, hipStream_t s)
{
    colstats_kernel<<<M, 256, 0, s>>>(X, NNODES, M, mean, var);
    size_t nm = (size_t)NNODES * M;
    bn_apply_kernel<<<cdiv(nm, 256), 256, 0, s>>>(X, mean, var, g, b, Y, NNODES, M, relu);
}

extern "C" void kernel_launch(void* const* d_in, const int* in_sizes, int n_in,
                              void* d_out, int out_size, void* d_ws, size_t ws_size,
                              hipStream_t stream)
{
    const int N = NNODES, E = NEDGES;

    // -------- inputs (setup_inputs dict order, params flattened in order) ---
    const float* x    = (const float*)d_in[0];
    const int*   ei   = (const int*)  d_in[1];   // [2, E]
    /* batch d_in[2] unused */
    const float* eps  = (const float*)d_in[3];
    const float* emb  = (const float*)d_in[4];
    GatP enc0{ (const float*)d_in[5],  (const float*)d_in[6],  (const float*)d_in[7]  };
    GatP enc1{ (const float*)d_in[9],  (const float*)d_in[10], (const float*)d_in[11] };
    GatP enc2{ (const float*)d_in[13], (const float*)d_in[14], (const float*)d_in[15] };
    const float* ebn0g = (const float*)d_in[17]; const float* ebn0b = (const float*)d_in[18];
    const float* ebn1g = (const float*)d_in[19]; const float* ebn1b = (const float*)d_in[20];
    const float* ebn2g = (const float*)d_in[21]; const float* ebn2b = (const float*)d_in[22];
    const float* zmW = (const float*)d_in[23]; const float* zmb = (const float*)d_in[24];
    const float* zlW = (const float*)d_in[25]; const float* zlb = (const float*)d_in[26];
    GatP dec0{ (const float*)d_in[27], (const float*)d_in[28], (const float*)d_in[29] };
    GatP dec1{ (const float*)d_in[31], (const float*)d_in[32], (const float*)d_in[33] };
    GatP dec2{ (const float*)d_in[35], (const float*)d_in[36], (const float*)d_in[37] };
    const float* dec2b = (const float*)d_in[38];
    const float* dbn0g = (const float*)d_in[39]; const float* dbn0b = (const float*)d_in[40];
    const float* dbn1g = (const float*)d_in[41]; const float* dbn1b = (const float*)d_in[42];
    const float* clsW = (const float*)d_in[43]; const float* clsb = (const float*)d_in[44];
    const float* nb0W = (const float*)d_in[45]; const float* nb0b = (const float*)d_in[46];
    const float* nb1W = (const float*)d_in[47]; const float* nb1b = (const float*)d_in[48];
    const float* nb2W = (const float*)d_in[49]; const float* nb2b = (const float*)d_in[50];

    const int* srcp = ei;        // edge_index[0]
    const int* dstp = ei + E;    // edge_index[1]

    // -------- outputs: cont[N,128] | canid[N,512] | nb[N,512] | z[N,32] | kl
    float* out   = (float*)d_out;
    float* contO = out;
    float* canO  = out + (size_t)N * 128;
    float* nbO   = out + (size_t)N * 640;
    float* zO    = out + (size_t)N * 1152;
    float* klO   = out + (size_t)N * 1184;

    // -------- workspace carve ----------------------------------------------
    float* w = (float*)d_ws;
    float* h0   = w; w += (size_t)N * K0PAD;
    float* t1   = w; w += (size_t)N * 128;   // transformed features (hw)
    float* t2   = w; w += (size_t)N * 128;   // aggregation target
    float* t3   = w; w += (size_t)N * 128;   // BN output / next layer input
    float* ssrc = w; w += (size_t)N * 4;
    float* sdst = w; w += (size_t)N * 4;
    unsigned* amax = (unsigned*)w; w += (size_t)N * 4;
    float* den  = w; w += (size_t)N * 4;
    float* ex   = w; w += (size_t)EPRIME * 4;
    float* mu   = w; w += (size_t)N * 32;
    float* lv   = w; w += (size_t)N * 32;
    float* mean = w; w += 128;
    float* var  = w; w += 128;
    float* klacc = w; w += 3;                 // leaves wpad below 12B past 16B alignment
    float* wpad = w; w += (size_t)K0PAD * 128; // padded enc0 weight [160,128]

    // -------- encoder ------------------------------------------------------
    {
        size_t tot = (size_t)N * K0PAD;
        embed_concat_kernel<<<cdiv(tot, 256), 256, 0, stream>>>(x, emb, h0, N);
    }
    // enc0 weight: zero-pad rows 136..159 so K=160 GEMM never reads OOB.
    {
        size_t tot = (size_t)K0PAD * 128;
        fill_f32_kernel<<<cdiv(tot, 256), 256, 0, stream>>>(wpad, 0.f, tot);
        hipMemcpyAsync(wpad, enc0.W, (size_t)136 * 128 * sizeof(float),
                       hipMemcpyDeviceToDevice, stream);
    }
    // enc0: GAT(136(pad160) -> 4x32), BN, ReLU  (GAT bias is a no-op under BN)
    GatP enc0p{ wpad, enc0.as_, enc0.ad_ };
    run_gat(h0, K0PAD, 4, 32, enc0p, srcp, dstp, t1, t2, ssrc, sdst, amax, den, ex, stream);
    run_bn(t2, ebn0g, ebn0b, t3, 128, 1, mean, var, stream);
    // enc1: GAT(128 -> 1x32), BN, ReLU
    run_gat(t3, 128, 1, 32, enc1, srcp, dstp, t1, t2, ssrc, sdst, amax, den, ex, stream);
    run_bn(t2, ebn1g, ebn1b, t3, 32, 1, mean, var, stream);
    // enc2: GAT(32 -> 1x32), BN, ReLU
    run_gat(t3, 32, 1, 32, enc2, srcp, dstp, t1, t2, ssrc, sdst, amax, den, ex, stream);
    run_bn(t2, ebn2g, ebn2b, t3, 32, 1, mean, var, stream);

    // -------- latent -------------------------------------------------------
    run_gemm(t3, zmW, zmb, mu, 32, 32, 0, stream);
    run_gemm(t3, zlW, zlb, lv, 32, 32, 0, stream);
    fill_f32_kernel<<<1, 32, 0, stream>>>(klacc, 0.f, 1);
    {
        int tot = N * 32;
        z_kl_kernel<<<cdiv(tot, 256), 256, 0, stream>>>(mu, lv, eps, zO, klacc, tot);
    }

    // -------- decoder ------------------------------------------------------
    // dec0: GAT(32 -> 4x32), BN, ReLU
    run_gat(zO, 32, 4, 32, dec0, srcp, dstp, t1, t2, ssrc, sdst, amax, den, ex, stream);
    run_bn(t2, dbn0g, dbn0b, t3, 128, 1, mean, var, stream);
    // dec1: GAT(128 -> 4x32), BN, ReLU
    run_gat(t3, 128, 4, 32, dec1, srcp, dstp, t1, t2, ssrc, sdst, amax, den, ex, stream);
    run_bn(t2, dbn1g, dbn1b, t3, 128, 1, mean, var, stream);
    // dec2: GAT(128 -> 1x128), +bias, sigmoid -> cont
    run_gat(t3, 128, 1, 128, dec2, srcp, dstp, t1, t2, ssrc, sdst, amax, den, ex, stream);
    {
        size_t nm = (size_t)N * 128;
        bias_sigmoid_kernel<<<cdiv(nm, 256), 256, 0, stream>>>(t2, dec2b, contO, N, 128);
    }

    // -------- canid head: bn(cont, dbn1) @ clsW + clsb ---------------------
    run_bn(contO, dbn1g, dbn1b, t3, 128, 0, mean, var, stream);
    run_gemm(t3, clsW, clsb, canO, 128, 512, 0, stream);

    // -------- neighborhood MLP: z -> 32 -> 32 -> 512 -----------------------
    run_gemm(zO, nb0W, nb0b, t1, 32, 32, 1, stream);
    run_gemm(t1, nb1W, nb1b, t2, 32, 32, 1, stream);
    run_gemm(t2, nb2W, nb2b, nbO, 32, 512, 0, stream);

    // -------- KL scalar ----------------------------------------------------
    kl_finalize_kernel<<<1, 32, 0, stream>>>(klacc, klO, N * 32);
}
